// ModulatedDeformablePSRoIPooling_42262478193269
// MI455X (gfx1250) — matrix operand order repaired and checked by reading, not compiled, not run on hardware
//
#include <hip/hip_runtime.h>
#include <hip/hip_bf16.h>
#include <math.h>

// ---------------------------------------------------------------------------
// Types for CDNA5 WMMA (wave32, 16x16x32 bf16 -> f32)
// ---------------------------------------------------------------------------
typedef __attribute__((ext_vector_type(16))) __bf16 bf16x16;
typedef __attribute__((ext_vector_type(8)))  __bf16 bf16x8;
typedef __attribute__((ext_vector_type(8)))  float  f32x8;

#define POOLED 7
#define SR 2
#define SCALE 0.0625f
#define TRANS_STD 0.1f

#define BATCH 2
#define CCH 256
#define HH 160
#define WW 160
#define NROI 1024
#define FC 1024
#define DIN (CCH * POOLED * POOLED)   // 12544
#define PP (POOLED * POOLED)          // 49
#define NTAP 16                       // 4 subsamples x 4 bilinear corners

// ---------------------------------------------------------------------------
// LDS-tiled transpose + f32 -> bf16 downcast.  src [R, C] f32 -> dst [C, R] bf16
// ---------------------------------------------------------------------------
__global__ __launch_bounds__(256) void transpose_to_bf16(
    const float* __restrict__ src, __bf16* __restrict__ dst, int R, int C)
{
    __shared__ float tile[32][33];
    int cb = blockIdx.x * 32;
    int rb = blockIdx.y * 32;
    #pragma unroll
    for (int i = threadIdx.y; i < 32; i += 8) {
        int r = rb + i, c = cb + threadIdx.x;
        tile[i][threadIdx.x] = (r < R && c < C) ? src[(size_t)r * C + c] : 0.0f;
    }
    __syncthreads();
    #pragma unroll
    for (int i = threadIdx.y; i < 32; i += 8) {
        int c = cb + i, r = rb + threadIdx.x;
        if (c < C && r < R)
            dst[(size_t)c * R + r] = (__bf16)tile[threadIdx.x][i];
    }
}

// ---------------------------------------------------------------------------
// Pool prep: resolve each (roi, bin) into 16 (offset, weight) taps.
// Weights absorb bilinear coeffs, 1/cnt normalization, and (pass 2) the
// sigmoid mask.  f3 == nullptr -> zero offsets, mask = 1 (pass 1).
// ---------------------------------------------------------------------------
__global__ __launch_bounds__(256) void pool_prep(
    const float* __restrict__ rois, const float* __restrict__ f3,
    int* __restrict__ toff, float* __restrict__ twgt)
{
    int t = blockIdx.x * 256 + threadIdx.x;
    if (t >= NROI * PP) return;
    int n = t / PP, s = t - n * PP;
    int ph = s / POOLED, pw = s - ph * POOLED;

    const float* r = rois + n * 5;
    float x1 = floorf(r[1] + 0.5f) * SCALE - 0.5f;
    float y1 = floorf(r[2] + 0.5f) * SCALE - 0.5f;
    float x2 = (floorf(r[3] + 0.5f) + 1.0f) * SCALE - 0.5f;
    float y2 = (floorf(r[4] + 0.5f) + 1.0f) * SCALE - 0.5f;
    float roi_w = fmaxf(x2 - x1, 0.1f);
    float roi_h = fmaxf(y2 - y1, 0.1f);
    float bin_w = roi_w * (1.0f / POOLED);
    float bin_h = roi_h * (1.0f / POOLED);
    float sub_w = bin_w * (1.0f / SR);
    float sub_h = bin_h * (1.0f / SR);

    float tx = 0.0f, ty = 0.0f, mask = 1.0f;
    if (f3) {
        const float* fr = f3 + (size_t)n * (3 * PP);
        tx = fr[s] * TRANS_STD;
        ty = fr[PP + s] * TRANS_STD;
        mask = 1.0f / (1.0f + expf(-fr[2 * PP + s]));
    }
    float wstart = (float)pw * bin_w + x1 + tx * roi_w;
    float hstart = (float)ph * bin_h + y1 + ty * roi_h;

    int   off[NTAP];
    float wg[NTAP];
    int cnt = 0;
    #pragma unroll
    for (int iy = 0; iy < SR; ++iy) {
        #pragma unroll
        for (int ix = 0; ix < SR; ++ix) {
            int j = (iy * SR + ix) * 4;
            float w = wstart + (float)ix * sub_w;
            float h = hstart + (float)iy * sub_h;
            bool valid = !(w < -0.5f || w > (float)WW - 0.5f ||
                           h < -0.5f || h > (float)HH - 0.5f);
            float wc = fminf(fmaxf(w, 0.0f), (float)WW - 1.0f);
            float hc = fminf(fmaxf(h, 0.0f), (float)HH - 1.0f);
            int w0 = (int)floorf(wc);
            int h0 = (int)floorf(hc);
            int w1 = (int)ceilf(wc);
            int h1 = (int)ceilf(hc);
            float dx = wc - (float)w0;
            float dy = hc - (float)h0;
            float v = valid ? 1.0f : 0.0f;
            off[j + 0] = valid ? (h0 * WW + w0) : 0;
            off[j + 1] = valid ? (h0 * WW + w1) : 0;
            off[j + 2] = valid ? (h1 * WW + w0) : 0;
            off[j + 3] = valid ? (h1 * WW + w1) : 0;
            wg[j + 0] = v * (1.0f - dx) * (1.0f - dy);
            wg[j + 1] = v * dx * (1.0f - dy);
            wg[j + 2] = v * (1.0f - dx) * dy;
            wg[j + 3] = v * dx * dy;
            cnt += valid ? 1 : 0;
        }
    }
    float sc = (cnt > 0) ? mask / (float)cnt : 0.0f;
    #pragma unroll
    for (int j = 0; j < NTAP; ++j) {
        toff[(size_t)t * NTAP + j] = off[j];
        twgt[(size_t)t * NTAP + j] = wg[j] * sc;
    }
}

// ---------------------------------------------------------------------------
// Pool apply: one block per RoI, taps staged in LDS (broadcast reads),
// one thread per channel.  Writes either bf16 (pass 1: WMMA A-matrix) or
// f32 (pass 2: final output) -- indices coincide: n*CCH*PP + c*PP + s.
// ---------------------------------------------------------------------------
__global__ __launch_bounds__(256) void pool_apply(
    const float* __restrict__ data, const float* __restrict__ rois,
    const int* __restrict__ toff, const float* __restrict__ twgt,
    __bf16* __restrict__ outb, float* __restrict__ outf)
{
    __shared__ int   soff[PP * NTAP];
    __shared__ float swgt[PP * NTAP];
    int n = blockIdx.x;
    int c = threadIdx.x;
    for (int i = threadIdx.x; i < PP * NTAP; i += 256) {
        soff[i] = toff[(size_t)n * (PP * NTAP) + i];
        swgt[i] = twgt[(size_t)n * (PP * NTAP) + i];
    }
    __syncthreads();

    int b = (int)rois[n * 5];
    const float* img = data + ((size_t)b * CCH + c) * (HH * WW);
    size_t obase = (size_t)n * (CCH * PP) + (size_t)c * PP;

    if (outb) {
        for (int s = 0; s < PP; ++s) {
            float acc = 0.0f;
            #pragma unroll
            for (int j = 0; j < NTAP; ++j)
                acc += swgt[s * NTAP + j] * img[soff[s * NTAP + j]];
            outb[obase + s] = (__bf16)acc;
        }
    } else {
        for (int s = 0; s < PP; ++s) {
            float acc = 0.0f;
            #pragma unroll
            for (int j = 0; j < NTAP; ++j)
                acc += swgt[s * NTAP + j] * img[soff[s * NTAP + j]];
            outf[obase + s] = acc;
        }
    }
}

// ---------------------------------------------------------------------------
// WMMA GEMM (split-K partial): part[z] = A[:, k0:k0+klen] * BT[:, k0:k0+klen]^T
// block = 256 threads = 8 waves (2 M x 4 N); wave computes 64x64 via 4x4 WMMA
// tiles.  16 MAC/byte of fragment traffic; operands stream from L2, with
// near-cache (WGP-scope) prefetch 16 k-steps ahead.
// ---------------------------------------------------------------------------
__global__ __launch_bounds__(256) void gemm_bf16_wmma_partial(
    const __bf16* __restrict__ A,    // [M, K] row-major
    const __bf16* __restrict__ BT,   // [N, K] row-major (B transposed)
    float* __restrict__ part,        // [Z, M, N] f32 partials
    int M, int N, int K, int klen)
{
    const int lane = threadIdx.x & 31;
    const int wid  = threadIdx.x >> 5;
    const int wm = wid & 1;          // 0..1
    const int wn = wid >> 1;         // 0..3
    const int m0 = blockIdx.x * 128 + wm * 64;
    const int n0 = blockIdx.y * 256 + wn * 64;
    const int k0 = blockIdx.z * klen;
    const int kend = k0 + klen;
    const int half = lane >> 4;
    const int l16  = lane & 15;

    f32x8 acc[4][4];
    #pragma unroll
    for (int mt = 0; mt < 4; ++mt)
        #pragma unroll
        for (int nt = 0; nt < 4; ++nt)
            #pragma unroll
            for (int r = 0; r < 8; ++r) acc[mt][nt][r] = 0.0f;

    const __bf16* arow[4];
    const __bf16* brow[4];
    #pragma unroll
    for (int i = 0; i < 4; ++i) {
        arow[i] = A  + (size_t)(m0 + i * 16 + l16) * K;
        brow[i] = BT + (size_t)(n0 + i * 16 + l16) * K;
    }

    for (int kk = k0; kk < kend; kk += 32) {
        bf16x16 afrag[4], bfrag[4];
        #pragma unroll
        for (int mt = 0; mt < 4; ++mt) {
            // A lane layout: contiguous K = half*16 + 0..15
            bf16x8 lo = *(const bf16x8*)(arow[mt] + kk + half * 16);
            bf16x8 hi = *(const bf16x8*)(arow[mt] + kk + half * 16 + 8);
            afrag[mt] = __builtin_shufflevector(lo, hi,
                0, 1, 2, 3, 4, 5, 6, 7, 8, 9, 10, 11, 12, 13, 14, 15);
            // near-cache (WGP-scope) prefetch, 16 k-steps ahead
            __builtin_prefetch(arow[mt] + kk + 512, 0, 3);
        }
        #pragma unroll
        for (int nt = 0; nt < 4; ++nt) {
            // B lane layout: lane = column n, contiguous K = half*16 + 0..15
            bfrag[nt] = *(const bf16x16*)(brow[nt] + kk + half * 16);
            __builtin_prefetch(brow[nt] + kk + 512, 0, 3);
        }
        #pragma unroll
        for (int mt = 0; mt < 4; ++mt)
            #pragma unroll
            for (int nt = 0; nt < 4; ++nt)
                acc[mt][nt] = __builtin_amdgcn_wmma_f32_16x16x32_bf16(
                    false, afrag[mt], false, bfrag[nt],
                    (short)0, acc[mt][nt], false, false);
    }

    float* pz = part + (size_t)blockIdx.z * M * N;
    #pragma unroll
    for (int mt = 0; mt < 4; ++mt)
        #pragma unroll
        for (int nt = 0; nt < 4; ++nt)
            #pragma unroll
            for (int r = 0; r < 8; ++r) {
                int m = m0 + mt * 16 + half * 8 + r;
                int n = n0 + nt * 16 + l16;
                pz[(size_t)m * N + n] = acc[mt][nt][r];
            }
}

// ---------------------------------------------------------------------------
// Split-K reduce + bias + ReLU + downcast to bf16 (deterministic, no atomics)
// ---------------------------------------------------------------------------
__global__ __launch_bounds__(256) void reduce_relu_bf16(
    const float* __restrict__ part, int Z, const float* __restrict__ bias,
    __bf16* __restrict__ out, int M, int N)
{
    int i = blockIdx.x * 256 + threadIdx.x;
    if (i >= M * N) return;
    int n = i & (N - 1);   // N is a power of two here
    float v = bias[n];
    for (int z = 0; z < Z; ++z) v += part[(size_t)z * M * N + i];
    out[i] = (__bf16)fmaxf(v, 0.0f);
}

// ---------------------------------------------------------------------------
// Tiny third layer: f3 = relu(f2[1024,1024]bf16 @ w3[1024,147]f32 + b3) -> f32
// ---------------------------------------------------------------------------
__global__ __launch_bounds__(160) void gemm3_relu(
    const __bf16* __restrict__ f2, const float* __restrict__ w3,
    const float* __restrict__ b3, float* __restrict__ f3)
{
    int n = blockIdx.x;
    int j = threadIdx.x;
    if (j >= 3 * PP) return;
    const __bf16* fr = f2 + (size_t)n * FC;
    float acc = b3[j];
    #pragma unroll 4
    for (int k = 0; k < FC; ++k)
        acc += (float)fr[k] * w3[k * (3 * PP) + j];
    f3[n * (3 * PP) + j] = fmaxf(acc, 0.0f);
}

// ---------------------------------------------------------------------------
// Launch
// ---------------------------------------------------------------------------
extern "C" void kernel_launch(void* const* d_in, const int* in_sizes, int n_in,
                              void* d_out, int out_size, void* d_ws, size_t ws_size,
                              hipStream_t stream)
{
    const float* data = (const float*)d_in[0];  // [2,256,160,160]
    const float* rois = (const float*)d_in[1];  // [1024,5]
    const float* w1   = (const float*)d_in[2];  // [12544,1024]
    const float* b1   = (const float*)d_in[3];  // [1024]
    const float* w2   = (const float*)d_in[4];  // [1024,1024]
    const float* b2   = (const float*)d_in[5];  // [1024]
    const float* w3   = (const float*)d_in[6];  // [1024,147]
    const float* b3   = (const float*)d_in[7];  // [147]
    float* out = (float*)d_out;                 // [1024,256,7,7]

    char* ws = (char*)d_ws;
    size_t off = 0;
    auto take = [&](size_t bytes) { char* p = ws + off; off += (bytes + 255) & ~(size_t)255; return p; };

    __bf16* Abf  = (__bf16*)take((size_t)NROI * DIN * 2);        // pooled features bf16
    __bf16* wt1  = (__bf16*)take((size_t)FC * DIN * 2);          // w1^T bf16 [FC, DIN]
    __bf16* wt2  = (__bf16*)take((size_t)FC * FC * 2);           // w2^T bf16 [FC, FC]
    float*  part = (float*) take((size_t)4 * NROI * FC * 4);     // split-K partials (Z<=4)
    __bf16* f1   = (__bf16*)take((size_t)NROI * FC * 2);
    __bf16* f2   = (__bf16*)take((size_t)NROI * FC * 2);
    float*  f3   = (float*) take((size_t)NROI * 3 * PP * 4);
    int*    toff = (int*)   take((size_t)NROI * PP * NTAP * 4);  // tap offsets
    float*  twgt = (float*) take((size_t)NROI * PP * NTAP * 4);  // tap weights
    (void)ws_size; (void)in_sizes; (void)n_in; (void)out_size;

    // 1) weight transpose + downcast
    transpose_to_bf16<<<dim3((FC + 31) / 32, (DIN + 31) / 32), dim3(32, 8), 0, stream>>>(
        w1, wt1, DIN, FC);
    transpose_to_bf16<<<dim3((FC + 31) / 32, (FC + 31) / 32), dim3(32, 8), 0, stream>>>(
        w2, wt2, FC, FC);

    // 2) pooling pass 1: prep taps (zero offsets) then apply -> bf16 A matrix
    pool_prep<<<(NROI * PP + 255) / 256, 256, 0, stream>>>(rois, nullptr, toff, twgt);
    pool_apply<<<NROI, 256, 0, stream>>>(data, rois, toff, twgt, Abf, nullptr);

    // 3) GEMM1: [1024 x 12544] x [12544 x 1024], split-K=4, then reduce+relu
    gemm_bf16_wmma_partial<<<dim3(NROI / 128, FC / 256, 4), 256, 0, stream>>>(
        Abf, wt1, part, NROI, FC, DIN, DIN / 4);
    reduce_relu_bf16<<<(NROI * FC + 255) / 256, 256, 0, stream>>>(
        part, 4, b1, f1, NROI, FC);

    // 4) GEMM2: [1024 x 1024] x [1024 x 1024], split-K=2, then reduce+relu
    gemm_bf16_wmma_partial<<<dim3(NROI / 128, FC / 256, 2), 256, 0, stream>>>(
        f1, wt2, part, NROI, FC, FC, FC / 2);
    reduce_relu_bf16<<<(NROI * FC + 255) / 256, 256, 0, stream>>>(
        part, 2, b2, f2, NROI, FC);

    // 5) GEMM3 (tiny, VALU): -> f3 [1024, 147]
    gemm3_relu<<<NROI, 160, 0, stream>>>(f2, w3, b3, f3);

    // 6) pooling pass 2: prep with learned offsets + sigmoid mask, apply -> f32 out
    pool_prep<<<(NROI * PP + 255) / 256, 256, 0, stream>>>(rois, f3, toff, twgt);
    pool_apply<<<NROI, 256, 0, stream>>>(data, rois, toff, twgt, nullptr, out);
}
